// GCN_90795608637581
// MI455X (gfx1250) — matrix operand order repaired and checked by reading, not compiled, not run on hardware
//
#include <hip/hip_runtime.h>

typedef __attribute__((ext_vector_type(16))) _Float16 v16h;
typedef __attribute__((ext_vector_type(8)))  _Float16 v8h;
typedef __attribute__((ext_vector_type(8)))  float    v8f;

#define CH 128
#define EPS 1e-5f

// ---------------- small utility kernels ----------------

__global__ void k_init_deg(float* __restrict__ deg, int n) {
    int i = blockIdx.x * blockDim.x + threadIdx.x;
    if (i < n) deg[i] = 1.0f;   // self-loop
}

__global__ void k_edge_deg(const int* __restrict__ dst, float* __restrict__ deg, int e) {
    int i = blockIdx.x * blockDim.x + threadIdx.x;
    if (i < e) atomicAdd(&deg[dst[i]], 1.0f);
}

__global__ void k_dinv(const float* __restrict__ deg, float* __restrict__ dinv, int n) {
    int i = blockIdx.x * blockDim.x + threadIdx.x;
    if (i < n) dinv[i] = rsqrtf(deg[i]);
}

// per-edge symmetric norm, computed once, used by both scatter passes
__global__ void k_norm(const int* __restrict__ src, const int* __restrict__ dst,
                       const float* __restrict__ dinv, float* __restrict__ norm, int e) {
    int i = blockIdx.x * blockDim.x + threadIdx.x;
    if (i < e) norm[i] = dinv[src[i]] * dinv[dst[i]];
}

// Pack W[128][128] f32 into WMMA B-fragment order:
// Wp[((w*4 + ks)*32 + lane)*16 + j] = (f16) W[(ks*32 + (lane>>4)*16 + j)*128 + w*16 + (lane&15)]
__global__ void k_pack_w(const float* __restrict__ W, _Float16* __restrict__ Wp) {
    int idx = blockIdx.x * blockDim.x + threadIdx.x;   // 0..16383
    if (idx >= CH * CH) return;
    int j    = idx & 15;
    int lane = (idx >> 4) & 31;
    int ks   = (idx >> 9) & 3;
    int w    = idx >> 11;
    int kk   = ks * 32 + (lane >> 4) * 16 + j;
    int col  = w * 16 + (lane & 15);
    Wp[idx] = (_Float16)W[kk * CH + col];
}

// ---------------- shared WMMA core ----------------
// xs: 16x128 f16 tile in LDS. Wave w computes 16x16 output tile at cols [16w,16w+16).
__device__ __forceinline__ v8f wmma_tile(const _Float16* xs, const v16h* Bfrag,
                                         int wave, int lane) {
    int lr = lane & 15;
    int hi = lane >> 4;
    v8f acc = {};
    #pragma unroll
    for (int ks = 0; ks < 4; ++ks) {
        int k0 = ks * 32;
        v16h a;
        #pragma unroll
        for (int i = 0; i < 8; ++i) {
            int kk = k0 + hi * 16 + 2 * i;
            a[2 * i]     = xs[lr * CH + kk];
            a[2 * i + 1] = xs[lr * CH + kk + 1];
        }
        v16h b = Bfrag[(wave * 4 + ks) * 32 + lane];   // 32B coalesced
        acc = __builtin_amdgcn_wmma_f32_16x16x32_f16(false, a, false, b,
                                                     (short)0, acc, false, false);
    }
    return acc;
}

// ---------------- GEMM (plain): H = X @ W ----------------
__global__ __launch_bounds__(256) void k_gemm_wmma(const float* __restrict__ X,
                                                   const _Float16* __restrict__ Wp,
                                                   float* __restrict__ H, int n) {
    __shared__ __align__(16) _Float16 xs[16 * CH];
    int row0 = blockIdx.x * 16;
    {
        int i0 = threadIdx.x * 8;
        int r  = row0 + (i0 >> 7);
        r = r < n ? r : n - 1;
        const float4* xp = (const float4*)(X + (long)r * CH + (i0 & 127));
        float4 f0 = xp[0], f1 = xp[1];
        v8h hv = { (_Float16)f0.x, (_Float16)f0.y, (_Float16)f0.z, (_Float16)f0.w,
                   (_Float16)f1.x, (_Float16)f1.y, (_Float16)f1.z, (_Float16)f1.w };
        *(v8h*)(xs + i0) = hv;
    }
    __syncthreads();

    int wave = threadIdx.x >> 5;
    int lane = threadIdx.x & 31;
    v8f acc = wmma_tile(xs, (const v16h*)Wp, wave, lane);

    int lr = lane & 15, hi = lane >> 4;
    int col = wave * 16 + lr;
    if (row0 + 16 <= n) {                       // uniform fast path: no masking
        #pragma unroll
        for (int j = 0; j < 8; ++j)
            H[(long)(row0 + j + hi * 8) * CH + col] = acc[j];
    } else {
        #pragma unroll
        for (int j = 0; j < 8; ++j) {
            int row = row0 + j + hi * 8;
            if (row < n) H[(long)row * CH + col] = acc[j];
        }
    }
}

// ---------------- GEMM (fused layer 2): H2 = relu(T*scale+shift) @ W ;
//                  OUT = H2*dinv^2 + b2  (written from registers) ----------------
__global__ __launch_bounds__(256) void k_gemm_bn_out(const float* __restrict__ T,
                                                     const float* __restrict__ scale,
                                                     const float* __restrict__ shift,
                                                     const _Float16* __restrict__ Wp,
                                                     const float* __restrict__ dinv,
                                                     const float* __restrict__ b2,
                                                     float* __restrict__ H,
                                                     float* __restrict__ OUT, int n) {
    __shared__ __align__(16) _Float16 xs[16 * CH];
    __shared__ float sdinv[16];
    int row0 = blockIdx.x * 16;
    if (threadIdx.x < 16) {
        int r = row0 + threadIdx.x;
        r = r < n ? r : n - 1;
        float di = dinv[r];
        sdinv[threadIdx.x] = di * di;
    }
    {
        int i0 = threadIdx.x * 8;
        int r  = row0 + (i0 >> 7);
        r = r < n ? r : n - 1;
        int c0 = i0 & 127;
        const float4* xp = (const float4*)(T + (long)r * CH + c0);
        float4 f0 = xp[0], f1 = xp[1];
        float4 s0 = *(const float4*)(scale + c0), s1 = *(const float4*)(scale + c0 + 4);
        float4 t0 = *(const float4*)(shift + c0), t1 = *(const float4*)(shift + c0 + 4);
        f0.x = fmaxf(f0.x * s0.x + t0.x, 0.f);
        f0.y = fmaxf(f0.y * s0.y + t0.y, 0.f);
        f0.z = fmaxf(f0.z * s0.z + t0.z, 0.f);
        f0.w = fmaxf(f0.w * s0.w + t0.w, 0.f);
        f1.x = fmaxf(f1.x * s1.x + t1.x, 0.f);
        f1.y = fmaxf(f1.y * s1.y + t1.y, 0.f);
        f1.z = fmaxf(f1.z * s1.z + t1.z, 0.f);
        f1.w = fmaxf(f1.w * s1.w + t1.w, 0.f);
        v8h hv = { (_Float16)f0.x, (_Float16)f0.y, (_Float16)f0.z, (_Float16)f0.w,
                   (_Float16)f1.x, (_Float16)f1.y, (_Float16)f1.z, (_Float16)f1.w };
        *(v8h*)(xs + i0) = hv;
    }
    __syncthreads();

    int wave = threadIdx.x >> 5;
    int lane = threadIdx.x & 31;
    v8f acc = wmma_tile(xs, (const v16h*)Wp, wave, lane);

    int lr = lane & 15, hi = lane >> 4;
    int col = wave * 16 + lr;
    float bc = b2[col];
    if (row0 + 16 <= n) {
        #pragma unroll
        for (int j = 0; j < 8; ++j) {
            int rj = j + hi * 8;
            long idx = (long)(row0 + rj) * CH + col;
            float v = acc[j];
            H[idx]   = v;                        // raw h2 for edge scatter
            OUT[idx] = v * sdinv[rj] + bc;       // self-loop + bias init
        }
    } else {
        #pragma unroll
        for (int j = 0; j < 8; ++j) {
            int rj = j + hi * 8;
            int row = row0 + rj;
            if (row < n) {
                long idx = (long)row * CH + col;
                float v = acc[j];
                H[idx]   = v;
                OUT[idx] = v * sdinv[rj] + bc;
            }
        }
    }
}

// ---------------- edge scatter: AGG[dst] += H[src] * norm[e] ----------------
__global__ __launch_bounds__(256) void k_scatter(const float* __restrict__ H,
                                                 const float* __restrict__ norm,
                                                 const int* __restrict__ src,
                                                 const int* __restrict__ dst,
                                                 float* __restrict__ AGG, int e) {
    int eidx = blockIdx.x * 8 + (threadIdx.x >> 5);
    if (eidx >= e) return;
    int lane = threadIdx.x & 31;
    int s = src[eidx], d = dst[eidx];
    float nrm = norm[eidx];
    const float4 v = ((const float4*)(H + (long)s * CH))[lane];
    float* ap = AGG + (long)d * CH + lane * 4;
    atomicAdd(ap + 0, v.x * nrm);
    atomicAdd(ap + 1, v.y * nrm);
    atomicAdd(ap + 2, v.z * nrm);
    atomicAdd(ap + 3, v.w * nrm);
}

// ---------------- layer-1 epilogue (float4): t = agg + h*dinv^2 + b1, BN stats ----
__global__ __launch_bounds__(256) void k_finish1(float4* __restrict__ agg,
                                                 const float4* __restrict__ h,
                                                 const float* __restrict__ dinv,
                                                 const float* __restrict__ b,
                                                 float* __restrict__ sums,
                                                 float* __restrict__ sumsq, int n) {
    long tid = (long)blockIdx.x * blockDim.x + threadIdx.x;
    long stride = (long)gridDim.x * blockDim.x;     // float4 units; *4 multiple of 128
    int c0 = (int)((tid * 4) & 127);
    float4 bc = *(const float4*)(b + c0);
    float4 s = {0.f, 0.f, 0.f, 0.f}, q = {0.f, 0.f, 0.f, 0.f};
    long total4 = (long)n * (CH / 4);
    for (long i = tid; i < total4; i += stride) {
        int node = (int)(i >> 5);
        float di = dinv[node];
        float di2 = di * di;
        float4 a = agg[i], hh = h[i], v;
        v.x = a.x + hh.x * di2 + bc.x;
        v.y = a.y + hh.y * di2 + bc.y;
        v.z = a.z + hh.z * di2 + bc.z;
        v.w = a.w + hh.w * di2 + bc.w;
        agg[i] = v;
        s.x += v.x; s.y += v.y; s.z += v.z; s.w += v.w;
        q.x += v.x * v.x; q.y += v.y * v.y; q.z += v.z * v.z; q.w += v.w * v.w;
    }
    atomicAdd(&sums[c0 + 0], s.x);  atomicAdd(&sums[c0 + 1], s.y);
    atomicAdd(&sums[c0 + 2], s.z);  atomicAdd(&sums[c0 + 3], s.w);
    atomicAdd(&sumsq[c0 + 0], q.x); atomicAdd(&sumsq[c0 + 1], q.y);
    atomicAdd(&sumsq[c0 + 2], q.z); atomicAdd(&sumsq[c0 + 3], q.w);
}

__global__ void k_bnparams(const float* __restrict__ sums, const float* __restrict__ sumsq,
                           const float* __restrict__ gamma, const float* __restrict__ beta,
                           float* __restrict__ scale, float* __restrict__ shift, int n) {
    int c = threadIdx.x;
    if (c < CH) {
        float invn = 1.0f / (float)n;
        float mean = sums[c] * invn;
        float var  = sumsq[c] * invn - mean * mean;
        float sc = gamma[c] * rsqrtf(var + EPS);
        scale[c] = sc;
        shift[c] = beta[c] - mean * sc;
    }
}

// ---------------- host-side orchestration ----------------

extern "C" void kernel_launch(void* const* d_in, const int* in_sizes, int n_in,
                              void* d_out, int out_size, void* d_ws, size_t ws_size,
                              hipStream_t stream) {
    const float* x      = (const float*)d_in[0];
    const float* W1     = (const float*)d_in[1];
    const float* b1     = (const float*)d_in[2];
    const float* gamma1 = (const float*)d_in[3];
    const float* beta1  = (const float*)d_in[4];
    const float* W2     = (const float*)d_in[5];
    const float* b2     = (const float*)d_in[6];
    const int*   ei     = (const int*)d_in[7];

    int N = in_sizes[0] / CH;
    int E = in_sizes[7] / 2;
    const int* src = ei;
    const int* dst = ei + E;
    float* out = (float*)d_out;

    char* ws = (char*)d_ws;
    size_t off = 0;
    auto carve = [&](size_t bytes) -> void* {
        void* p = ws + off;
        off = (off + bytes + 255) & ~(size_t)255;
        return p;
    };
    float*    h      = (float*)carve((size_t)N * CH * 4);   // h1 then h2
    float*    t      = (float*)carve((size_t)N * CH * 4);   // agg1 -> out1 (raw, pre-BN)
    float*    deg    = (float*)carve((size_t)N * 4);
    float*    dinv   = (float*)carve((size_t)N * 4);
    float*    norm   = (float*)carve((size_t)E * 4);
    _Float16* W1p    = (_Float16*)carve(CH * CH * 2);
    _Float16* W2p    = (_Float16*)carve(CH * CH * 2);
    float*    sums   = (float*)carve(CH * 4);
    float*    sumsq  = (float*)carve(CH * 4);
    float*    scale  = (float*)carve(CH * 4);
    float*    shift  = (float*)carve(CH * 4);

    int gN   = (N + 255) / 256;
    int gE   = (E + 255) / 256;
    int gEw  = (E + 7) / 8;
    int gRow = (N + 15) / 16;

    k_pack_w<<<(CH * CH + 255) / 256, 256, 0, stream>>>(W1, W1p);
    k_pack_w<<<(CH * CH + 255) / 256, 256, 0, stream>>>(W2, W2p);

    k_init_deg<<<gN, 256, 0, stream>>>(deg, N);
    k_edge_deg<<<gE, 256, 0, stream>>>(dst, deg, E);
    k_dinv<<<gN, 256, 0, stream>>>(deg, dinv, N);
    k_norm<<<gE, 256, 0, stream>>>(src, dst, dinv, norm, E);

    // ---- layer 1 ----
    hipMemsetAsync(t, 0, (size_t)N * CH * 4, stream);
    hipMemsetAsync(sums, 0, CH * 4, stream);
    hipMemsetAsync(sumsq, 0, CH * 4, stream);
    k_gemm_wmma<<<gRow, 256, 0, stream>>>(x, W1p, h, N);
    k_scatter<<<gEw, 256, 0, stream>>>(h, norm, src, dst, t, E);
    k_finish1<<<2048, 256, 0, stream>>>((float4*)t, (const float4*)h, dinv, b1,
                                        sums, sumsq, N);
    k_bnparams<<<1, 128, 0, stream>>>(sums, sumsq, gamma1, beta1, scale, shift, N);

    // ---- layer 2 (BN+ReLU fused into GEMM input; out-init fused into epilogue) ----
    k_gemm_bn_out<<<gRow, 256, 0, stream>>>(t, scale, shift, W2p, dinv, b2, h, out, N);
    k_scatter<<<gEw, 256, 0, stream>>>(h, norm, src, dst, out, E);
}
